// S4Model_28166395527871
// MI455X (gfx1250) — compile-verified, matrix operand order checked by reading, not run on hardware
//
#include <hip/hip_runtime.h>
#include <hip/hip_bf16.h>
#include <math.h>

typedef __bf16 bf16_t;
typedef __attribute__((ext_vector_type(16))) __bf16 v16bf;
typedef __attribute__((ext_vector_type(8)))  float  v8f;

#define B_    32
#define DIN_  64
#define L_    4096
#define H_    256
#define N2_   32
#define NL_   4
#define Q_    64                 // SSM chunk length
#define NCH_  (L_/Q_)            // 64 chunks per row
#define NCOL_ (B_*L_)
#define LN_EPS_ 1e-5f

__device__ __forceinline__ float gelu_f(float x) {
    return 0.5f * x * (1.f + erff(x * 0.70710678118654752f));
}
__device__ __forceinline__ float sigmoid_f(float x) {
    return 1.f / (1.f + expf(-x));
}
__device__ __forceinline__ v8f zero8() {
    v8f z;
#pragma unroll
    for (int i = 0; i < 8; ++i) z[i] = 0.f;
    return z;
}
// Per-lane slice of a 16x32 bf16 WMMA fragment. p = rowbase + kh*8 (kh=lane>>4):
// elements 0..7 = K{0..7}+kh*8, elements 8..15 = K{16..23}+kh*8 (ISA 16-bit layout).
__device__ __forceinline__ v16bf ld_frag(const bf16_t* p) {
    v16bf f;
#pragma unroll
    for (int i = 0; i < 8; ++i) { f[i] = p[i]; f[i + 8] = p[i + 16]; }
    return f;
}
// Async global->LDS 16B copy (ASYNCcnt). lds_dst must point into __shared__
// memory: the low 32 bits of the generic address are the DS-relative offset
// (generic shared ptr = {aperture_hi, lds_offset}); hardware adds LDS_BASE.
// Passing the pointer through ptrtoint also captures the shared object, so the
// compiler keeps the LDS allocation alive and honors the memory clobber.
__device__ __forceinline__ void async_cp16(bf16_t* lds_dst, const bf16_t* gsrc) {
    unsigned ldsoff = (unsigned)(unsigned long long)lds_dst;
    asm volatile("global_load_async_to_lds_b128 %0, %1, off"
                 :: "v"(ldsoff), "v"(gsrc) : "memory");
}
__device__ __forceinline__ void async_wait0() {
    asm volatile("s_wait_asynccnt 0x0" ::: "memory");
}

// ---------------- per-(layer,h,n): abar = exp(dtA), w = 2*C*(abar-1)/A --------
__global__ void k_prep_params(const float* __restrict__ log_dt,
                              const float* __restrict__ Cri,
                              const float* __restrict__ logAre,
                              const float* __restrict__ Aim,
                              float4* __restrict__ par) {
    int idx = blockIdx.x * blockDim.x + threadIdx.x;
    if (idx >= NL_ * H_ * N2_) return;
    int h  = (idx >> 5) & (H_ - 1);
    int ly = idx >> 13;
    float dt  = expf(log_dt[ly * H_ + h]);
    float Are = -expf(logAre[idx]);
    float Ai  = Aim[idx];
    float er  = expf(Are * dt);
    float abr = er * cosf(Ai * dt);
    float abi = er * sinf(Ai * dt);
    float cr = Cri[idx * 2 + 0], ci = Cri[idx * 2 + 1];
    float nr = abr - 1.f, ni = abi;
    float tr = cr * nr - ci * ni;
    float ti = cr * ni + ci * nr;
    float den = Are * Are + Ai * Ai;
    float ctr = (tr * Are + ti * Ai) / den;
    float cti = (ti * Are - tr * Ai) / den;
    par[idx] = make_float4(abr, abi, 2.f * ctr, 2.f * cti);
}

// ---------------- bf16 weights (encoder transposed, GLU row-major) ------------
__global__ void k_prep_weights(const float* __restrict__ enc_w,
                               const float* __restrict__ glu_w,
                               bf16_t* __restrict__ encT,
                               bf16_t* __restrict__ gluwb) {
    int idx = blockIdx.x * blockDim.x + threadIdx.x;
    if (idx < H_ * DIN_) {
        int d = idx >> 6, c = idx & 63;
        encT[idx] = (bf16_t)enc_w[c * H_ + d];
    }
    int i2 = idx - H_ * DIN_;
    if (i2 >= 0 && i2 < NL_ * 2 * H_ * H_) {
        gluwb[i2] = (bf16_t)glu_w[i2];
    }
}

// ---------------- chunked-SSM coefficient tables ------------------------------
// Wave per (layer,h); lane = state n. Iterates abar^m, w*abar^m for m=0..63:
//   E1[n][63-m]    = Re(abar^m), E1[32+n][63-m] = Im(abar^m)   (state accumulation)
//   KT[m]          = sum_n Re(w*abar^m)                        (conv taps)
//   TE2[m][64+n]   = Re(w*abar^{m+1}), TE2[m][96+n] = -Im(...) (state->output)
//   D4[n]          = abar^64                                   (chunk decay)
__global__ void k_prep_taps(const float4* __restrict__ PAR,
                            float* __restrict__ KT,
                            bf16_t* __restrict__ E1M,
                            bf16_t* __restrict__ TE2M,
                            float2* __restrict__ D4) {
    int wv   = (blockIdx.x << 3) + (threadIdx.x >> 5);   // (layer*H + h)
    int lane = threadIdx.x & 31;
    float4 p = PAR[wv * N2_ + lane];
    float ar = p.x, ai = p.y, wr = p.z, wi = p.w;
    float pr = 1.f, pi = 0.f;        // abar^m
    float cr = wr, ci = wi;          // w * abar^m
    bf16_t* E1  = E1M  + (size_t)wv * Q_ * Q_;
    bf16_t* TE2 = TE2M + (size_t)wv * Q_ * 2 * Q_;
    for (int m = 0; m < Q_; ++m) {
        E1[lane * Q_ + (Q_ - 1 - m)]        = (bf16_t)pr;
        E1[(32 + lane) * Q_ + (Q_ - 1 - m)] = (bf16_t)pi;
        float red = cr;
#pragma unroll
        for (int off = 16; off > 0; off >>= 1) red += __shfl_xor(red, off, 32);
        if (lane == 0) KT[wv * Q_ + m] = red;
        float ncr = cr * ar - ci * ai, nci = cr * ai + ci * ar;
        cr = ncr; ci = nci;
        float npr = pr * ar - pi * ai, npi = pr * ai + pi * ar;
        pr = npr; pi = npi;
        TE2[m * 128 + 64 + lane] = (bf16_t)cr;
        TE2[m * 128 + 96 + lane] = (bf16_t)(-ci);
    }
    D4[wv * N2_ + lane] = make_float2(pr, pi);
}

// Fill lower-triangular Toeplitz T half of [T | E2]: TE2[i][j] = k[i-j] (j<=i).
__global__ void k_fill_T(const float* __restrict__ KT, bf16_t* __restrict__ TE2M) {
    int idx = blockIdx.x * 256 + threadIdx.x;   // (lh, i, j)
    int j  = idx & 63;
    int i  = (idx >> 6) & 63;
    int lh = idx >> 12;
    TE2M[(size_t)lh * 8192 + i * 128 + j] =
        (j <= i) ? (bf16_t)KT[lh * Q_ + (i - j)] : (bf16_t)0.f;
}

// ---------------- (B,C,L) fp32 -> (B,L,C) bf16 transpose; mode1: skip+GeLU ----
__global__ void k_transpose_bf16(const float* __restrict__ in,
                                 const float* __restrict__ u,
                                 const float* __restrict__ dsk,
                                 bf16_t* __restrict__ outp,
                                 int C, int mode) {
    __shared__ float t[32][33];
    int blocksPerB = (C >> 5) * (L_ >> 5);
    int b  = blockIdx.x / blocksPerB;
    int r  = blockIdx.x % blocksPerB;
    int c0 = (r / (L_ >> 5)) << 5;
    int l0 = (r % (L_ >> 5)) << 5;
    int tx = threadIdx.x, ty = threadIdx.y;
#pragma unroll
    for (int j = 0; j < 4; ++j) {
        int c = c0 + ty + 8 * j;
        size_t off = ((size_t)b * C + c) * L_ + l0 + tx;
        float v = in[off];
        if (mode) {
            v += dsk[c] * u[off];
            v = gelu_f(v);
        }
        t[ty + 8 * j][tx] = v;
    }
    __syncthreads();
#pragma unroll
    for (int j = 0; j < 4; ++j) {
        int l = l0 + ty + 8 * j;
        outp[((size_t)b * L_ + l) * C + c0 + tx] = (bf16_t)t[tx][ty + 8 * j];
    }
}

// ---------------- encoder GEMM -> H0 fp32 + H0b bf16 --------------------------
__global__ void k_gemm_enc(const bf16_t* __restrict__ encT,
                           const bf16_t* __restrict__ xT,
                           const float* __restrict__ enc_b,
                           float* __restrict__ H0,
                           bf16_t* __restrict__ H0b) {
    __shared__ bf16_t As[16 * DIN_];    // 2KB A tile shared by all 8 waves
    int tid = threadIdx.x;
    int mt = blockIdx.x & 15;
    int cb = blockIdx.x >> 4;
    if (tid < 128) {                    // 16*64 bf16 = 128 x 16B
        int row = tid >> 3, koff = (tid & 7) * 8;
        async_cp16(As + tid * 8,
                   encT + ((size_t)(mt * 16 + row)) * DIN_ + koff);
    }
    async_wait0();
    __syncthreads();
    int wave = tid >> 5, lane = tid & 31;
    int gc = (cb * 8 + wave) * 16;
    int b  = gc >> 12;
    int l0 = gc & (L_ - 1);
    int laneM = lane & 15, kh = lane >> 4;
    const bf16_t* Ar = As + laneM * DIN_ + kh * 8;
    const bf16_t* Br = xT + ((size_t)b * L_ + l0 + laneM) * DIN_ + kh * 8;
    v8f acc = zero8();
#pragma unroll
    for (int k0 = 0; k0 < DIN_; k0 += 32) {
        v16bf a  = ld_frag(Ar + k0);
        v16bf bb = ld_frag(Br + k0);
        acc = __builtin_amdgcn_wmma_f32_16x16x32_bf16(false, a, false, bb,
                                                      (short)0, acc, false, false);
    }
#pragma unroll
    for (int r = 0; r < 8; ++r) {
        int m = mt * 16 + r + kh * 8;
        int l = l0 + laneM;
        float v = acc[r] + enc_b[m];
        size_t off = ((size_t)b * H_ + m) * L_ + l;
        H0[off]  = v;
        H0b[off] = (bf16_t)v;
    }
}

// ---------------- P = E1 @ U : in-chunk state accumulation --------------------
// Grid: h(256) x mt(4) x cb(16); block covers 1 mtile x 128 columns (cols=(b,c)).
__global__ void k_gemm_p(const bf16_t* __restrict__ E1,   // (H,64,64) layer slice
                         const bf16_t* __restrict__ Ub,   // H0b (B,H,L)
                         float* __restrict__ P) {         // (H,B,NCH,64)
    __shared__ bf16_t As[16 * Q_];
    int tid = threadIdx.x;
    int h  = blockIdx.x >> 6;
    int mt = (blockIdx.x >> 4) & 3;
    int cb = blockIdx.x & 15;
    if (tid < 128) {
        int row = tid >> 3, koff = (tid & 7) * 8;
        async_cp16(As + tid * 8,
                   E1 + ((size_t)h * Q_ + mt * 16 + row) * Q_ + koff);
    }
    async_wait0();
    __syncthreads();
    int wave = tid >> 5, lane = tid & 31;
    int laneM = lane & 15, kh = lane >> 4;
    int col = (cb * 8 + wave) * 16 + laneM;     // (b,c) column in [0, B*NCH)
    int b = col >> 6, c = col & 63;
    const bf16_t* Ar = As + laneM * Q_ + kh * 8;
    const bf16_t* Br = Ub + ((size_t)b * H_ + h) * L_ + c * Q_ + kh * 8;
    v8f acc = zero8();
#pragma unroll
    for (int k0 = 0; k0 < Q_; k0 += 32) {
        v16bf a  = ld_frag(Ar + k0);
        v16bf bb = ld_frag(Br + k0);
        acc = __builtin_amdgcn_wmma_f32_16x16x32_bf16(false, a, false, bb,
                                                      (short)0, acc, false, false);
    }
    size_t cbase = (((size_t)h * B_ + b) * NCH_ + c) * 64;
#pragma unroll
    for (int r = 0; r < 8; ++r)
        P[cbase + mt * 16 + r + kh * 8] = acc[r];
}

// ---------------- inter-chunk recurrence: s[c] = abar^Q * s[c-1] + p[c] -------
// Thread per (h,b,n); stores state BEFORE each chunk (bf16 GEMM operand).
__global__ void k_chunk_scan(const float2* __restrict__ D4,  // (H,N2) layer slice
                             const float* __restrict__ P,
                             bf16_t* __restrict__ S) {       // (H,B,NCH,64)
    int idx = blockIdx.x * 256 + threadIdx.x;
    int n  = idx & 31;
    int hb = idx >> 5;           // h*B + b
    int h  = hb >> 5;
    float2 d = D4[h * N2_ + n];
    float dr = d.x, di = d.y;
    float sr = 0.f, si = 0.f;
    size_t base = (size_t)hb * NCH_ * 64;
    for (int c = 0; c < NCH_; ++c) {
        size_t o = base + c * 64;
        S[o + n]      = (bf16_t)sr;
        S[o + 32 + n] = (bf16_t)si;
        float pre = P[o + n], pim = P[o + 32 + n];
        float nr = dr * sr - di * si + pre;
        float ni = dr * si + di * sr + pim;
        sr = nr; si = ni;
    }
}

// ---------------- Y = [T | E2] @ [U ; S] : intra conv + state->output ---------
__global__ void k_gemm_y(const bf16_t* __restrict__ TE2,  // (H,64,128) layer slice
                         const bf16_t* __restrict__ Ub,   // H0b
                         const bf16_t* __restrict__ S,    // (H,B,NCH,64)
                         float* __restrict__ Y) {         // H1 (B,H,L)
    __shared__ bf16_t As[16 * 2 * Q_];   // 4KB
    int tid = threadIdx.x;
    int h  = blockIdx.x >> 6;
    int mt = (blockIdx.x >> 4) & 3;
    int cb = blockIdx.x & 15;
    {
        int row = tid >> 4, koff = (tid & 15) * 8;    // 16x128 bf16 = 256 x 16B
        async_cp16(As + tid * 8,
                   TE2 + ((size_t)h * Q_ + mt * 16 + row) * 128 + koff);
    }
    async_wait0();
    __syncthreads();
    int wave = tid >> 5, lane = tid & 31;
    int laneM = lane & 15, kh = lane >> 4;
    int col = (cb * 8 + wave) * 16 + laneM;
    int b = col >> 6, c = col & 63;
    const bf16_t* Ar = As + laneM * 128 + kh * 8;
    const bf16_t* Bu = Ub + ((size_t)b * H_ + h) * L_ + c * Q_ + kh * 8;
    const bf16_t* Bs = S + (((size_t)h * B_ + b) * NCH_ + c) * 64 + kh * 8;
    v8f acc = zero8();
#pragma unroll
    for (int k0 = 0; k0 < Q_; k0 += 32) {              // intra-chunk conv part
        v16bf a  = ld_frag(Ar + k0);
        v16bf bb = ld_frag(Bu + k0);
        acc = __builtin_amdgcn_wmma_f32_16x16x32_bf16(false, a, false, bb,
                                                      (short)0, acc, false, false);
    }
#pragma unroll
    for (int k0 = 0; k0 < Q_; k0 += 32) {              // state decay part
        v16bf a  = ld_frag(Ar + Q_ + k0);
        v16bf bb = ld_frag(Bs + k0);
        acc = __builtin_amdgcn_wmma_f32_16x16x32_bf16(false, a, false, bb,
                                                      (short)0, acc, false, false);
    }
#pragma unroll
    for (int r = 0; r < 8; ++r) {
        int i = mt * 16 + r + kh * 8;   // position within chunk
        Y[((size_t)b * H_ + h) * L_ + c * Q_ + i] = acc[r];
    }
}

// ---------------- GLU GEMM with async-staged weights + fused epilogue ---------
__global__ void k_gemm_glu(const bf16_t* __restrict__ W,   // (512,256) layer slice
                           const bf16_t* __restrict__ TB,  // (B,L,H) bf16
                           const float* __restrict__ gb,   // glu_b (512)
                           const float* __restrict__ res,  // H0
                           float* __restrict__ outp) {     // H1
    __shared__ bf16_t As[32 * H_];      // 16KB: 16 a-rows + 16 g-rows
    int tid = threadIdx.x;
    int mt = blockIdx.x & 15;
    int cb = blockIdx.x >> 4;
#pragma unroll
    for (int jj = 0; jj < 4; ++jj) {    // 32*256 bf16 = 1024 x 16B
        int q = tid + 256 * jj;
        int row  = q >> 5;
        int koff = (q & 31) * 8;
        int grow = (row < 16) ? (mt * 16 + row) : (H_ + mt * 16 + row - 16);
        async_cp16(As + q * 8, W + (size_t)grow * H_ + koff);
    }
    async_wait0();
    __syncthreads();
    int wave = tid >> 5, lane = tid & 31;
    int gc = (cb * 8 + wave) * 16;
    int b  = gc >> 12;
    int l0 = gc & (L_ - 1);
    int laneM = lane & 15, kh = lane >> 4;
    const bf16_t* Aa = As + laneM * H_ + kh * 8;
    const bf16_t* Ag = As + (16 + laneM) * H_ + kh * 8;
    const bf16_t* Br = TB + ((size_t)b * L_ + l0 + laneM) * H_ + kh * 8;
    v8f acca = zero8(), accg = zero8();
#pragma unroll
    for (int k0 = 0; k0 < H_; k0 += 32) {
        v16bf bb = ld_frag(Br + k0);
        v16bf a0 = ld_frag(Aa + k0);
        v16bf a1 = ld_frag(Ag + k0);
        acca = __builtin_amdgcn_wmma_f32_16x16x32_bf16(false, a0, false, bb,
                                                       (short)0, acca, false, false);
        accg = __builtin_amdgcn_wmma_f32_16x16x32_bf16(false, a1, false, bb,
                                                       (short)0, accg, false, false);
    }
#pragma unroll
    for (int r = 0; r < 8; ++r) {
        int o = mt * 16 + r + kh * 8;
        int l = l0 + laneM;
        float av = acca[r] + gb[o];
        float gv = accg[r] + gb[H_ + o];
        size_t off = ((size_t)b * H_ + o) * L_ + l;
        outp[off] = av * sigmoid_f(gv) + res[off];
    }
}

// ---------------- LayerNorm over channels -> H0 fp32 + H0b bf16 ---------------
__global__ void k_layernorm(const float* __restrict__ in,   // H1
                            const float* __restrict__ g,
                            const float* __restrict__ be,
                            float* __restrict__ outp,       // H0
                            bf16_t* __restrict__ outb) {    // H0b
    int b = blockIdx.x >> 4;
    int l = ((blockIdx.x & 15) << 8) + threadIdx.x;
    size_t base = (size_t)b * H_ * L_ + l;
    float s = 0.f, s2 = 0.f;
    for (int c = 0; c < H_; ++c) {
        float v = in[base + (size_t)c * L_];
        s += v; s2 += v * v;
    }
    float mu  = s * (1.f / H_);
    float var = s2 * (1.f / H_) - mu * mu;
    float rstd = rsqrtf(var + LN_EPS_);
    for (int c = 0; c < H_; ++c) {
        float v = in[base + (size_t)c * L_];
        float o = (v - mu) * rstd * g[c] + be[c];
        outp[base + (size_t)c * L_] = o;
        outb[base + (size_t)c * L_] = (bf16_t)o;
    }
}

// ---------------- mean over time ----------------------------------------------
__global__ void k_pool(const float* __restrict__ h, float* __restrict__ pooled) {
    int wave = (blockIdx.x << 3) + (threadIdx.x >> 5);
    int lane = threadIdx.x & 31;
    const float* p = h + (size_t)wave * L_;
    float s = 0.f;
    for (int i = 0; i < L_ / 32; ++i) s += p[lane + 32 * i];
#pragma unroll
    for (int off = 16; off > 0; off >>= 1) s += __shfl_xor(s, off, 32);
    if (lane == 0) pooled[wave] = s * (1.f / L_);
}

// ---------------- decoder + softmax -------------------------------------------
__global__ void k_head(const float* __restrict__ pooled,
                       const float* __restrict__ dw,
                       const float* __restrict__ db,
                       float* __restrict__ outp) {
    int b = threadIdx.x;
    if (b >= B_) return;
    float l0 = db[0], l1 = db[1];
    for (int hh = 0; hh < H_; ++hh) {
        float p = pooled[b * H_ + hh];
        l0 = fmaf(p, dw[hh * 2 + 0], l0);
        l1 = fmaf(p, dw[hh * 2 + 1], l1);
    }
    float m = fmaxf(l0, l1);
    float e0 = expf(l0 - m), e1 = expf(l1 - m);
    float inv = 1.f / (e0 + e1);
    outp[b * 2 + 0] = e0 * inv;
    outp[b * 2 + 1] = e1 * inv;
}

extern "C" void kernel_launch(void* const* d_in, const int* in_sizes, int n_in,
                              void* d_out, int out_size, void* d_ws, size_t ws_size,
                              hipStream_t stream) {
    const float* x      = (const float*)d_in[0];
    const float* enc_w  = (const float*)d_in[1];
    const float* enc_b  = (const float*)d_in[2];
    const float* log_dt = (const float*)d_in[3];
    const float* Cri    = (const float*)d_in[4];
    const float* logAre = (const float*)d_in[5];
    const float* Aim    = (const float*)d_in[6];
    const float* Dskip  = (const float*)d_in[7];
    const float* glu_w  = (const float*)d_in[8];
    const float* glu_b  = (const float*)d_in[9];
    const float* ln_g   = (const float*)d_in[10];
    const float* ln_b   = (const float*)d_in[11];
    const float* dec_w  = (const float*)d_in[12];
    const float* dec_b  = (const float*)d_in[13];
    float* out = (float*)d_out;

    char* ws = (char*)d_ws;
    size_t off = 0;
    auto alloc = [&](size_t bytes) -> char* {
        char* p = ws + off;
        off += (bytes + 255) & ~(size_t)255;
        return p;
    };
    float*  H0   = (float*) alloc((size_t)NCOL_ * H_ * 4);        // h fp32
    float*  H1   = (float*) alloc((size_t)NCOL_ * H_ * 4);        // scratch fp32
    bf16_t* H0b  = (bf16_t*)alloc((size_t)NCOL_ * H_ * 2);        // h bf16 (B,H,L)
    bf16_t* TB   = (bf16_t*)alloc((size_t)NCOL_ * H_ * 2);        // (B,L,H) bf16
    float*  Pb   = (float*) alloc((size_t)H_ * B_ * NCH_ * 64 * 4);
    bf16_t* Sb   = (bf16_t*)alloc((size_t)H_ * B_ * NCH_ * 64 * 2);
    float4* PAR  = (float4*)alloc((size_t)NL_ * H_ * N2_ * 16);
    float*  KT   = (float*) alloc((size_t)NL_ * H_ * Q_ * 4);
    bf16_t* E1M  = (bf16_t*)alloc((size_t)NL_ * H_ * Q_ * Q_ * 2);
    bf16_t* TE2M = (bf16_t*)alloc((size_t)NL_ * H_ * Q_ * 2 * Q_ * 2);
    float2* D4   = (float2*)alloc((size_t)NL_ * H_ * N2_ * 8);
    bf16_t* ENCT = (bf16_t*)alloc((size_t)H_ * DIN_ * 2);
    bf16_t* GLW  = (bf16_t*)alloc((size_t)NL_ * 2 * H_ * H_ * 2);
    float*  POOL = (float*) alloc((size_t)B_ * H_ * 4);

    dim3 tblk(32, 8);
    k_prep_params<<<(NL_ * H_ * N2_ + 255) / 256, 256, 0, stream>>>(log_dt, Cri, logAre, Aim, PAR);
    k_prep_weights<<<(H_ * DIN_ + NL_ * 2 * H_ * H_ + 255) / 256, 256, 0, stream>>>(enc_w, glu_w, ENCT, GLW);
    k_prep_taps<<<(NL_ * H_) / 8, 256, 0, stream>>>(PAR, KT, E1M, TE2M, D4);
    k_fill_T<<<(NL_ * H_ * Q_ * Q_) / 256, 256, 0, stream>>>(KT, TE2M);
    k_transpose_bf16<<<B_ * (DIN_ / 32) * (L_ / 32), tblk, 0, stream>>>(x, nullptr, nullptr, TB, DIN_, 0);
    k_gemm_enc<<<16 * (NCOL_ / 128), 256, 0, stream>>>(ENCT, TB, enc_b, H0, H0b);
    for (int ly = 0; ly < NL_; ++ly) {
        k_gemm_p<<<H_ * 4 * 16, 256, 0, stream>>>(E1M + (size_t)ly * H_ * Q_ * Q_, H0b, Pb);
        k_chunk_scan<<<(H_ * B_ * N2_) / 256, 256, 0, stream>>>(D4 + (size_t)ly * H_ * N2_, Pb, Sb);
        k_gemm_y<<<H_ * 4 * 16, 256, 0, stream>>>(TE2M + (size_t)ly * H_ * Q_ * 2 * Q_, H0b, Sb, H1);
        k_transpose_bf16<<<B_ * (H_ / 32) * (L_ / 32), tblk, 0, stream>>>(H1, H0, Dskip + ly * H_, TB, H_, 1);
        k_gemm_glu<<<16 * (NCOL_ / 128), 256, 0, stream>>>(GLW + (size_t)ly * 2 * H_ * H_, TB,
                                                           glu_b + ly * 2 * H_, H0, H1);
        k_layernorm<<<B_ * 16, 256, 0, stream>>>(H1, ln_g + ly * H_, ln_b + ly * H_, H0, H0b);
    }
    k_pool<<<(B_ * H_) / 8, 256, 0, stream>>>(H0, POOL);
    k_head<<<1, 32, 0, stream>>>(POOL, dec_w, dec_b, out);
}
